// NeuronCircuitQKV_9990093931274
// MI455X (gfx1250) — matrix operand order repaired and checked by reading, not compiled, not run on hardware
//
#include <hip/hip_runtime.h>
#include <stdint.h>

// ---------------- constants (fixed problem size) ----------------
// B=2, S=2048, D=1024, R=1024, H=8, DH=128, NI=8, NP=32, NO=8, K=3
#define B_  2
#define S_  2048
#define D_  1024
#define R_  1024
#define H_  8
#define DH_ 128
#define NI_ 8
#define NP_ 32
#define NO_ 8
#define TK_ 3
#define T_  (B_ * S_)   // 4096 tokens
// All circuit GEMMs are 1024 (K) x 1024 (C) with 8 experts.
#define KD_   1024
#define CD_   1024
#define NE_   8
#define KCB_  (KD_ / 32)   // 32 K-chunks
#define CT_   (CD_ / 16)   // 64 col tiles

typedef __attribute__((ext_vector_type(16))) __bf16   v16bf;
typedef __attribute__((ext_vector_type(8)))  __bf16   v8bf;
typedef __attribute__((ext_vector_type(8)))  float    v8f;
typedef __attribute__((ext_vector_type(4)))  uint32_t u32x4;
typedef __attribute__((ext_vector_type(8)))  int      i32x8;
typedef __attribute__((ext_vector_type(4)))  int      i32x4;

union V16U { v16bf v; v8bf h[2]; };

// ---------------- bf16 helpers (bit-exact RNE, no libcalls) ----------------
__device__ __forceinline__ __bf16 f2bf(float f) {
  uint32_t u = __builtin_bit_cast(uint32_t, f);
  u += 0x7FFFu + ((u >> 16) & 1u);
  uint16_t h = (uint16_t)(u >> 16);
  return __builtin_bit_cast(__bf16, h);
}

// WMMA 16-bit fragment K-index maps (CDNA5 ISA §7.12.2), simplified:
//   A 16x32: row = lane&15, k(e) = e + 8*khalf + (e>=8 ? 8 : 0)
//   B 32x16: col = lane&15, same k(e) as A (this file uses k = e + 16*khalf for A and B)
__device__ __forceinline__ int a_k_of(int e, int khalf) {
  return e + 8 * khalf + ((e >= 8) ? 8 : 0);
}

// =====================================================================
// Router: logits for NI(8) + NP(32) + NO(8) rows, softmax(in/out), top-3.
// =====================================================================
__global__ __launch_bounds__(256)
void router_kernel(const float* __restrict__ x,
                   const float* __restrict__ Wi,
                   const float* __restrict__ Wp,
                   const float* __restrict__ Wo,
                   float* __restrict__ in_w,
                   int*   __restrict__ pidx,
                   float* __restrict__ out_w,
                   int Din) {
  __shared__ float sx[1024];
  __shared__ float logits[NI_ + NP_ + NO_];
  const int t = blockIdx.x;
  const int tid = threadIdx.x;

  for (int i = tid; i < Din; i += blockDim.x) sx[i] = x[(size_t)t * Din + i];
  __syncthreads();

  if (tid < NI_ + NP_ + NO_) {
    const float* wrow;
    if (tid < NI_)            wrow = Wi + (size_t)tid * Din;
    else if (tid < NI_ + NP_) wrow = Wp + (size_t)(tid - NI_) * Din;
    else                      wrow = Wo + (size_t)(tid - NI_ - NP_) * Din;
    float s = 0.f;
    for (int i = 0; i < Din; ++i) s += sx[i] * wrow[i];
    logits[tid] = s;
  }
  __syncthreads();

  if (tid == 0) {                       // softmax over NI
    float m = -1e30f;
    for (int i = 0; i < NI_; ++i) m = fmaxf(m, logits[i]);
    float e[NI_], sum = 0.f;
    for (int i = 0; i < NI_; ++i) { e[i] = __expf(logits[i] - m); sum += e[i]; }
    for (int i = 0; i < NI_; ++i) in_w[(size_t)t * NI_ + i] = e[i] / sum;
  } else if (tid == 1) {                // softmax over NO
    float m = -1e30f;
    for (int i = 0; i < NO_; ++i) m = fmaxf(m, logits[NI_ + NP_ + i]);
    float e[NO_], sum = 0.f;
    for (int i = 0; i < NO_; ++i) { e[i] = __expf(logits[NI_ + NP_ + i] - m); sum += e[i]; }
    for (int i = 0; i < NO_; ++i) out_w[(size_t)t * NO_ + i] = e[i] / sum;
  } else if (tid == 2) {                // top-3 over NP (first-index tie break)
    float tv[NP_];
    for (int i = 0; i < NP_; ++i) tv[i] = logits[NI_ + i];
    for (int k = 0; k < TK_; ++k) {
      int bi = 0; float bv = tv[0];
      for (int i = 1; i < NP_; ++i) if (tv[i] > bv) { bv = tv[i]; bi = i; }
      pidx[(size_t)t * TK_ + k] = bi;
      tv[bi] = -1e30f;
    }
  }
}

// =====================================================================
// Pack fp32 weights (NE x 1024 x 1024) into bf16 WMMA B-fragment order:
//   Wp[((n*KCB + kcb)*CT + ct)*32*16 + lane*16 + e]
//     = bf16( W[n][kcb*32 + 16*(lane>>4) + e][ct*16 + (lane&15)] )
// One 32-byte contiguous block per lane per fragment.
// =====================================================================
__global__ __launch_bounds__(256)
void pack_weights_kernel(const float* __restrict__ W, __bf16* __restrict__ Wp) {
  const uint32_t i = blockIdx.x * 256u + threadIdx.x;   // 8M elements
  const int e    = i & 15;
  const int lane = (i >> 4) & 31;
  const int ct   = (i >> 9) & (CT_ - 1);
  const int kcb  = (i >> 15) & (KCB_ - 1);
  const int n    = i >> 20;
  const int k = kcb * 32 + 16 * (lane >> 4) + e;
  const int c = ct * 16 + (lane & 15);
  Wp[i] = f2bf(W[((size_t)n * KD_ + k) * CD_ + c]);
}

// =====================================================================
// Elementwise fp32 -> bf16 (row-major copy), 4-wide.
// =====================================================================
__global__ __launch_bounds__(256)
void f32_to_bf16_kernel(const float* __restrict__ src, __bf16* __restrict__ dst) {
  const uint32_t i = (blockIdx.x * 256u + threadIdx.x) * 4u;
  const float4 v = *(const float4*)(src + i);
  dst[i + 0] = f2bf(v.x);
  dst[i + 1] = f2bf(v.y);
  dst[i + 2] = f2bf(v.z);
  dst[i + 3] = f2bf(v.w);
}

// =====================================================================
// Expert-mixture GEMM:  out[t,c] = sum_n tokw[t,n] * sum_k act[t,k]*W[n,k,c]
// Block: 256 thr = 8 waves -> 16 rows x 128 cols.
// The 16x1024 fp32 activation tile is DMA'd into LDS by the Tensor Data
// Mover (TENSOR_LOAD_TO_LDS, tracked by TENSORcnt), then swizzled into
// bf16 WMMA A-fragment order in LDS. 8 per-expert accumulators; token
// weights applied in the epilogue via the C-layout row map. Inner loop:
// 8 WMMA per A ds-load-pair, B from pre-packed bf16 (contiguous 32B/lane).
// =====================================================================
__global__ __launch_bounds__(256)
void circuit_gemm_kernel(const float*  __restrict__ act,   // T x 1024
                         const __bf16* __restrict__ Wp,    // packed NE x 1024 x 1024
                         const float*  __restrict__ tokw,  // T x 8
                         float* __restrict__ out) {        // T x 1024
  __shared__ __attribute__((aligned(16))) float  sStage[16 * KD_];  // 64 KB (TDM dest)
  __shared__ __attribute__((aligned(32))) __bf16 sAct[16 * KD_];    // 32 KB
  __shared__ float sTokw[16 * NE_];
  const int tid   = threadIdx.x;
  const int wave  = tid >> 5;
  const int lane  = tid & 31;
  const int lrow  = lane & 15;
  const int khalf = lane >> 4;
  const int t0 = blockIdx.x * 16;
  const int ct = blockIdx.y * 8 + wave;          // global 16-col tile index

  // ---- TDM: async-copy the 16x1024 fp32 tile into LDS (wave 0 issues) ----
  if (wave == 0) {
    const uint64_t ga  = (uint64_t)(uintptr_t)(act + (size_t)t0 * KD_);
    const uint32_t lds = (uint32_t)(uintptr_t)&sStage[0];
    u32x4 g0;
    g0[0] = 1u;                                   // count=1, user descriptor
    g0[1] = lds;                                  // lds_addr
    g0[2] = (uint32_t)ga;                         // global_addr[31:0]
    g0[3] = (uint32_t)((ga >> 32) & 0x01FFFFFFu)  // global_addr[56:32]
            | (2u << 30);                         // type=2 ("image")
    i32x8 g1;
    g1[0] = (int)(2u << 16);                      // data_size=2 -> 4 bytes
    g1[1] = (int)((KD_ & 0xFFFFu) << 16);         // tensor_dim0[15:0] @ bit48
    g1[2] = (int)(((uint32_t)KD_ >> 16) |         // tensor_dim0[31:16]
                  (((uint32_t)T_ & 0xFFFFu) << 16)); // tensor_dim1[15:0]
    g1[3] = (int)(((uint32_t)T_ >> 16) |          // tensor_dim1[31:16]
                  ((uint32_t)KD_ << 16));         // tile_dim0 = 1024
    g1[4] = 16;                                   // tile_dim1 = 16 rows
    g1[5] = KD_;                                  // tensor_dim0_stride[31:0]
    g1[6] = 0;
    g1[7] = 0;
    i32x4 z4 = {0, 0, 0, 0};                      // groups 2/3: <=2-D tensor
    i32x8 z8 = {0, 0, 0, 0, 0, 0, 0, 0};
    __builtin_amdgcn_tensor_load_to_lds(g0, g1, z4, z4, z8, 0);
    __builtin_amdgcn_s_wait_tensorcnt(0);
  }
  for (int i = tid; i < 16 * NE_; i += 256)
    sTokw[i] = tokw[(size_t)(t0 + (i >> 3)) * NE_ + (i & 7)];
  __syncthreads();

  // ---- swizzle fp32 tile -> bf16 fragment order: sAct[(kcb*32+lane)*16+e] ----
  for (int i = tid; i < 16 * KD_; i += 256) {
    const int e  = i & 15;
    const int ln = (i >> 4) & 31;
    const int kc = i >> 9;
    const int row = ln & 15;
    const int k = kc * 32 + 16 * (ln >> 4) + e;
    sAct[i] = f2bf(sStage[row * KD_ + k]);
  }
  __syncthreads();

  const v8f zero8 = {0.f, 0.f, 0.f, 0.f, 0.f, 0.f, 0.f, 0.f};
  v8f acc[NE_];
#pragma unroll
  for (int n = 0; n < NE_; ++n) acc[n] = zero8;

  for (int kcb = 0; kcb < KCB_; ++kcb) {
    const v16bf a = *(const v16bf*)&sAct[(kcb * 32 + lane) * 16];
    const size_t fragBase = ((size_t)kcb * CT_ + ct) * (32 * 16) + lane * 16;
    if (kcb + 1 < KCB_)                                  // stream next chunk
      __builtin_prefetch(Wp + fragBase + (size_t)CT_ * (32 * 16), 0, 3);
#pragma unroll
    for (int n = 0; n < NE_; ++n) {
      const v16bf b = *(const v16bf*)(Wp + (size_t)n * (KD_ * CD_) + fragBase);
      acc[n] = __builtin_amdgcn_wmma_f32_16x16x32_bf16(false, a, false, b,
                                                       (short)0, acc[n], false, false);
    }
  }

#pragma unroll
  for (int j = 0; j < 8; ++j) {
    const int row = j + 8 * khalf;
    float r = 0.f;
#pragma unroll
    for (int n = 0; n < NE_; ++n) r += sTokw[row * NE_ + n] * acc[n][j];
    out[(size_t)(t0 + row) * CD_ + (ct * 16 + lrow)] = r;
  }
}

// =====================================================================
// Householder: h -= 2 v (h.v)/(v.v + 1e-8), 3 reflections, one blk/token
// =====================================================================
__global__ __launch_bounds__(256)
void householder_kernel(float* __restrict__ h,          // T x Rd
                        const float* __restrict__ vtab, // NP x Rd
                        const int* __restrict__ pidx,   // T x 3
                        int Rd) {
  __shared__ float redA[256], redB[256];
  const int t = blockIdx.x;
  const int tid = threadIdx.x;
  float* hrow = h + (size_t)t * Rd;

  float hv[4];
#pragma unroll
  for (int i = 0; i < 4; ++i) hv[i] = hrow[tid + i * 256];

  for (int kk = 0; kk < TK_; ++kk) {
    const int idx = pidx[(size_t)t * TK_ + kk];
    const float* vrow = vtab + (size_t)idx * Rd;
    float vl[4], vv = 0.f, hd = 0.f;
#pragma unroll
    for (int i = 0; i < 4; ++i) {
      float v = vrow[tid + i * 256];
      vl[i] = v; vv += v * v; hd += hv[i] * v;
    }
    redA[tid] = vv; redB[tid] = hd;
    __syncthreads();
    for (int s = 128; s > 0; s >>= 1) {
      if (tid < s) { redA[tid] += redA[tid + s]; redB[tid] += redB[tid + s]; }
      __syncthreads();
    }
    const float scale = 2.f * redB[0] / (redA[0] + 1e-8f);
    __syncthreads();
#pragma unroll
    for (int i = 0; i < 4; ++i) hv[i] -= scale * vl[i];
  }
#pragma unroll
  for (int i = 0; i < 4; ++i) hrow[tid + i * 256] = hv[i];
}

// =====================================================================
// Flash attention, causal. One wave per (b, h, 16-query tile).
// Q/K/V pre-converted to bf16 row-major; fragments are contiguous
// 16B/32B loads (no conversion VALU). 1/sqrt(DH) folded into softmax.
// =====================================================================
__global__ __launch_bounds__(32)
void attention_kernel(const __bf16* __restrict__ Q,
                      const __bf16* __restrict__ Km,
                      const __bf16* __restrict__ V,
                      float* __restrict__ O) {
  __shared__ float sP[16][32];
  __shared__ float sM[16], sL[16], sAl[16];

  const int lane  = threadIdx.x;
  const int lrow  = lane & 15;
  const int khalf = lane >> 4;

  int id = blockIdx.x;
  const int qt = id & (S_ / 16 - 1); id >>= 7;   // S_/16 = 128
  const int hh = id & (H_ - 1);
  const int bb = id >> 3;

  const size_t base = (size_t)bb * S_;
  const int    hoff = hh * DH_;
  const float  qscale = 0.088388347648318447f;   // 1/sqrt(128)

  // Q fragments: 16 x 128 in 4 chunks of K=32; two 16B loads each
  const __bf16* qrow = Q + (base + qt * 16 + lrow) * R_ + hoff;
  v16bf qa[4];
#pragma unroll
  for (int f = 0; f < 4; ++f) {
    V16U u;
    u.h[0] = *(const v8bf*)(qrow + f * 32 + 16 * khalf);
    u.h[1] = *(const v8bf*)(qrow + f * 32 + 16 * khalf + 8);
    qa[f] = u.v;
  }

  const v8f zero8 = {0.f, 0.f, 0.f, 0.f, 0.f, 0.f, 0.f, 0.f};
  v8f oacc[8];
#pragma unroll
  for (int f = 0; f < 8; ++f) oacc[f] = zero8;

  if (lane < 16) { sM[lane] = -1e30f; sL[lane] = 0.f; }
  __syncthreads();

  const int qend = qt * 16 + 15;

  for (int k0 = 0; k0 <= qend; k0 += 32) {
    // ---- scores for 32 keys (two 16-wide WMMA tiles) ----
#pragma unroll
    for (int half = 0; half < 2; ++half) {
      const int kc0 = k0 + half * 16;
      v8f sacc = zero8;
      if (kc0 <= qend) {
        const __bf16* krow = Km + (base + kc0 + lrow) * R_ + hoff;
#pragma unroll
        for (int f = 0; f < 4; ++f) {
          const v16bf kb = *(const v16bf*)(krow + f * 32 + 16 * khalf);
          sacc = __builtin_amdgcn_wmma_f32_16x16x32_bf16(false, qa[f], false, kb,
                                                         (short)0, sacc, false, false);
        }
      }
#pragma unroll
      for (int j = 0; j < 8; ++j) sP[j + 8 * khalf][half * 16 + lrow] = sacc[j];
    }
    __syncthreads();

    // ---- per-row online softmax (lanes 0..15 own one row each) ----
    if (lane < 16) {
      const int q = qt * 16 + lane;
      const float mold = sM[lane];
      float mnew = mold;
      float s[32];
#pragma unroll
      for (int c = 0; c < 32; ++c) {
        float sc = ((k0 + c) <= q) ? sP[lane][c] * qscale : -1e30f;
        s[c] = sc; mnew = fmaxf(mnew, sc);
      }
      float rsum = 0.f;
#pragma unroll
      for (int c = 0; c < 32; ++c) {
        float p = __expf(s[c] - mnew);
        sP[lane][c] = p; rsum += p;
      }
      const float alpha = __expf(mold - mnew);
      sAl[lane] = alpha;
      sM[lane]  = mnew;
      sL[lane]  = sL[lane] * alpha + rsum;
    }
    __syncthreads();

    // ---- rescale O and accumulate P @ V ----
    float alj[8];
#pragma unroll
    for (int j = 0; j < 8; ++j) alj[j] = sAl[j + 8 * khalf];

    v16bf pa;
#pragma unroll
    for (int e = 0; e < 16; ++e) pa[e] = f2bf(sP[lrow][16 * khalf + e]);

#pragma unroll
    for (int f = 0; f < 8; ++f) {
      v16bf vb;
#pragma unroll
      for (int e = 0; e < 16; ++e)
        vb[e] = V[(base + k0 + 16 * khalf + e) * R_ + hoff + f * 16 + lrow];
      v8f c = oacc[f];
#pragma unroll
      for (int j = 0; j < 8; ++j) c[j] *= alj[j];
      oacc[f] = __builtin_amdgcn_wmma_f32_16x16x32_bf16(false, pa, false, vb,
                                                        (short)0, c, false, false);
    }
    __syncthreads();
  }

  float linv[8];
#pragma unroll
  for (int j = 0; j < 8; ++j) linv[j] = 1.f / sL[j + 8 * khalf];
#pragma unroll
  for (int f = 0; f < 8; ++f)
#pragma unroll
    for (int j = 0; j < 8; ++j)
      O[(base + qt * 16 + j + 8 * khalf) * R_ + hoff + f * 16 + lrow] =
          oacc[f][j] * linv[j];
}

// =====================================================================
// Host launcher
// =====================================================================
extern "C" void kernel_launch(void* const* d_in, const int* in_sizes, int n_in,
                              void* d_out, int out_size, void* d_ws, size_t ws_size,
                              hipStream_t stream) {
  const float* x       = (const float*)d_in[0];
  // d_in[1] = causal mask (unused; causality handled analytically)
  const float* Wr_in   = (const float*)d_in[2];
  const float* Wr_proc = (const float*)d_in[3];
  const float* Wr_out  = (const float*)d_in[4];
  const float* WrO_in  = (const float*)d_in[5];
  const float* WrO_proc= (const float*)d_in[6];
  const float* WrO_out = (const float*)d_in[7];
  const float* Wq_in   = (const float*)d_in[8];
  const float* vq      = (const float*)d_in[9];
  const float* Wq_out  = (const float*)d_in[10];
  const float* Wk_in   = (const float*)d_in[11];
  const float* vk      = (const float*)d_in[12];
  const float* Wk_out  = (const float*)d_in[13];
  const float* Wv_in   = (const float*)d_in[14];
  const float* vv      = (const float*)d_in[15];
  const float* Wv_out  = (const float*)d_in[16];
  const float* Wo_in   = (const float*)d_in[17];
  const float* vo      = (const float*)d_in[18];
  const float* Wo_out  = (const float*)d_in[19];

  char* ws = (char*)d_ws;
  size_t off = 0;
  auto alloc = [&](size_t bytes) -> void* {
    void* p = ws + off;
    off += (bytes + 255) & ~(size_t)255;
    return p;
  };
  float*  in_w   = (float*)alloc((size_t)T_ * NI_ * 4);
  float*  out_w  = (float*)alloc((size_t)T_ * NO_ * 4);
  int*    pidx   = (int*)  alloc((size_t)T_ * TK_ * 4);
  float*  in_wO  = (float*)alloc((size_t)T_ * NI_ * 4);
  float*  out_wO = (float*)alloc((size_t)T_ * NO_ * 4);
  int*    pidxO  = (int*)  alloc((size_t)T_ * TK_ * 4);
  float*  hbuf   = (float*)alloc((size_t)T_ * 1024 * 4);
  float*  Qb     = (float*)alloc((size_t)T_ * R_ * 4);
  float*  Kb     = (float*)alloc((size_t)T_ * R_ * 4);
  float*  Vb     = (float*)alloc((size_t)T_ * R_ * 4);
  float*  ob     = (float*)alloc((size_t)T_ * R_ * 4);
  __bf16* wpack  = (__bf16*)alloc((size_t)NE_ * KD_ * CD_ * 2);   // 16 MB
  __bf16* Qbf    = (__bf16*)alloc((size_t)T_ * R_ * 2);
  __bf16* Kbf    = (__bf16*)alloc((size_t)T_ * R_ * 2);
  __bf16* Vbf    = (__bf16*)alloc((size_t)T_ * R_ * 2);

  const dim3 gemmGrid(T_ / 16, CD_ / 128);
  const int  packGrid = (NE_ * KD_ * CD_) / 256;        // 32768
  const int  cvtGrid  = ((size_t)T_ * R_) / (256 * 4);  // 4096

  // ---- input router ----
  router_kernel<<<T_, 256, 0, stream>>>(x, Wr_in, Wr_proc, Wr_out,
                                        in_w, pidx, out_w, D_);

  // ---- Q / K / V circuits ----
  pack_weights_kernel<<<packGrid, 256, 0, stream>>>(Wq_in, wpack);
  circuit_gemm_kernel<<<gemmGrid, 256, 0, stream>>>(x, wpack, in_w, hbuf);
  householder_kernel<<<T_, 256, 0, stream>>>(hbuf, vq, pidx, R_);
  pack_weights_kernel<<<packGrid, 256, 0, stream>>>(Wq_out, wpack);
  circuit_gemm_kernel<<<gemmGrid, 256, 0, stream>>>(hbuf, wpack, out_w, Qb);

  pack_weights_kernel<<<packGrid, 256, 0, stream>>>(Wk_in, wpack);
  circuit_gemm_kernel<<<gemmGrid, 256, 0, stream>>>(x, wpack, in_w, hbuf);
  householder_kernel<<<T_, 256, 0, stream>>>(hbuf, vk, pidx, R_);
  pack_weights_kernel<<<packGrid, 256, 0, stream>>>(Wk_out, wpack);
  circuit_gemm_kernel<<<gemmGrid, 256, 0, stream>>>(hbuf, wpack, out_w, Kb);

  pack_weights_kernel<<<packGrid, 256, 0, stream>>>(Wv_in, wpack);
  circuit_gemm_kernel<<<gemmGrid, 256, 0, stream>>>(x, wpack, in_w, hbuf);
  householder_kernel<<<T_, 256, 0, stream>>>(hbuf, vv, pidx, R_);
  pack_weights_kernel<<<packGrid, 256, 0, stream>>>(Wv_out, wpack);
  circuit_gemm_kernel<<<gemmGrid, 256, 0, stream>>>(hbuf, wpack, out_w, Vb);

  // ---- bf16 copies of Q/K/V for the attention matrix cores ----
  f32_to_bf16_kernel<<<cvtGrid, 256, 0, stream>>>(Qb, Qbf);
  f32_to_bf16_kernel<<<cvtGrid, 256, 0, stream>>>(Kb, Kbf);
  f32_to_bf16_kernel<<<cvtGrid, 256, 0, stream>>>(Vb, Vbf);

  // ---- causal attention ----
  attention_kernel<<<B_ * H_ * (S_ / 16), 32, 0, stream>>>(Qbf, Kbf, Vbf, ob);

  // ---- output router + output circuit ----
  router_kernel<<<T_, 256, 0, stream>>>(ob, WrO_in, WrO_proc, WrO_out,
                                        in_wO, pidxO, out_wO, R_);
  pack_weights_kernel<<<packGrid, 256, 0, stream>>>(Wo_in, wpack);
  circuit_gemm_kernel<<<gemmGrid, 256, 0, stream>>>(ob, wpack, in_wO, hbuf);
  householder_kernel<<<T_, 256, 0, stream>>>(hbuf, vo, pidxO, D_);
  pack_weights_kernel<<<packGrid, 256, 0, stream>>>(Wo_out, wpack);
  circuit_gemm_kernel<<<gemmGrid, 256, 0, stream>>>(hbuf, wpack, out_wO,
                                                    (float*)d_out);
}
